// TransformerBlock_64948495450212
// MI455X (gfx1250) — matrix operand order repaired and checked by reading, not compiled, not run on hardware
//
#include <hip/hip_runtime.h>
#include <hip/hip_bf16.h>
#include <math.h>
#include <stdint.h>

#define D_MODEL   1024
#define NUM_HEADS 16
#define DK        64
#define D_FF      4096
#define SEQ       2048
#define BATCH     4
#define NTOK      (BATCH * SEQ)

typedef __attribute__((ext_vector_type(16))) __bf16 v16bf;
typedef __attribute__((ext_vector_type(8)))  __bf16 v8bf;
typedef __attribute__((ext_vector_type(8)))  float  v8f;

// fp32 -> bf16, round-to-nearest-even
static __device__ __forceinline__ __bf16 f2bf(float f) {
  unsigned u = __builtin_bit_cast(unsigned, f);
  u += 0x7FFFu + ((u >> 16) & 1u);
  unsigned short h = (unsigned short)(u >> 16);
  return __builtin_bit_cast(__bf16, h);
}

static __device__ __forceinline__ void wait_async_le(int n) {
#if __has_builtin(__builtin_amdgcn_s_wait_asynccnt)
  if (n == 0) __builtin_amdgcn_s_wait_asynccnt(0);
  else        __builtin_amdgcn_s_wait_asynccnt(4);
#else
  if (n == 0) asm volatile("s_wait_asynccnt 0x0" ::: "memory");
  else        asm volatile("s_wait_asynccnt 0x4" ::: "memory");
#endif
}

// async 16B copy: global (per-lane 64-bit addr) -> LDS (per-lane byte addr)
static __device__ __forceinline__ void async_cp16(unsigned lds_addr,
                                                  const void* gptr) {
  uint64_t ga = (uint64_t)(uintptr_t)gptr;
  asm volatile("global_load_async_to_lds_b128 %0, %1, off"
               :: "v"(lds_addr), "v"(ga)
               : "memory");
}

__global__ void cvt_f32_to_bf16(const float* __restrict__ in,
                                __bf16* __restrict__ out, int n) {
  int i = blockIdx.x * blockDim.x + threadIdx.x;
  if (i < n) out[i] = f2bf(in[i]);
}

// ---------------------------------------------------------------------------
// C[M,N] = A[M,K] @ B[N,K]^T  (torch Linear; both operands K-contiguous)
// Block tile 128x128, double-buffered LDS fill via global_load_async_to_lds.
// 8 waves as 4(M) x 2(N); each wave: 32x64 tile = 8 WMMAs per 32-wide k-step.
// EPI: 0 = none, 1 = +bias, 2 = +bias then exact GELU.
// ---------------------------------------------------------------------------
#define BM    128
#define BN    128
#define BKK   32
#define ROWE  40                 // padded LDS row, elements (80B = 20 banks)
#define ROWB  (ROWE * 2)         // bytes
#define TILEB (BM * ROWB)        // 10240 B per tile

template <int EPI, bool OUTBF>
__global__ void __launch_bounds__(256)
gemm_wmma_bf16(const __bf16* __restrict__ A, const __bf16* __restrict__ B,
               const float* __restrict__ bias, float* __restrict__ Cf,
               __bf16* __restrict__ Cb, int M, int N, int K) {
  __shared__ __align__(128) unsigned char smem[4 * TILEB];  // A0 B0 A1 B1
  const int tid  = threadIdx.x;
  const int lane = tid & 31;
  const int wave = tid >> 5;
  const int half = lane >> 4;
  const int ln   = lane & 15;
  const int mw   = wave & 3;       // 4 waves along M, 32 rows each
  const int nw   = wave >> 2;      // 2 waves along N, 64 cols each
  const int m0 = blockIdx.y * BM;
  const int n0 = blockIdx.x * BN;

  const unsigned ldsBase = (unsigned)(uintptr_t)&smem[0];

  // Per stage, each thread async-copies 4 x 16B (2 for A, 2 for B):
  // chunk id c in [0,512): row = c/4 (128 rows), sub = c%4 (4 x 8 elems of k)
  auto issue = [&](int step, int buf) {
    const int k0 = step * BKK;
#pragma unroll
    for (int q = 0; q < 2; ++q) {
      const int cid = tid + q * 256;
      const int row = cid >> 2, sub = cid & 3;
      const unsigned lofs = (unsigned)(row * ROWB + sub * 16);
      async_cp16(ldsBase + buf * 2 * TILEB + lofs,
                 A + (size_t)(m0 + row) * K + k0 + sub * 8);
      async_cp16(ldsBase + buf * 2 * TILEB + TILEB + lofs,
                 B + (size_t)(n0 + row) * K + k0 + sub * 8);
    }
  };

  v8f zero = {};
  v8f acc[2][4];
#pragma unroll
  for (int s = 0; s < 2; ++s)
#pragma unroll
    for (int c = 0; c < 4; ++c) acc[s][c] = zero;

  const int steps = K / BKK;
  issue(0, 0);
  for (int i = 0; i < steps; ++i) {
    const int cur = i & 1;
    if (i + 1 < steps) {           // uniform branch
      issue(i + 1, cur ^ 1);
      wait_async_le(4);            // in-order: stage i's 4 copies done
    } else {
      wait_async_le(0);
    }
    __syncthreads();               // whole tile (all waves) resident

    const __bf16* tA = (const __bf16*)(smem + cur * 2 * TILEB);
    const __bf16* tB = (const __bf16*)(smem + cur * 2 * TILEB + TILEB);

    // A fragments (ISA 7.12.2): lanes 0-15 K=[0..7]+[16..23], 16-31 +8
    v16bf afr[2];
#pragma unroll
    for (int s = 0; s < 2; ++s) {
      const __bf16* ar = tA + (size_t)(mw * 32 + s * 16 + ln) * ROWE;
      v8bf lo = *(const v8bf*)(ar + half * 8);
      v8bf hi = *(const v8bf*)(ar + 16 + half * 8);
#pragma unroll
      for (int t = 0; t < 8; ++t) { afr[s][t] = lo[t]; afr[s][8 + t] = hi[t]; }
    }
#pragma unroll
    for (int c = 0; c < 4; ++c) {
      // B fragment: lanes 0-15 K=[0..7]+[16..23], 16-31 +8
      const __bf16* br = tB + (size_t)(nw * 64 + c * 16 + ln) * ROWE;
      v8bf blo = *(const v8bf*)(br + half * 8);
      v8bf bhi = *(const v8bf*)(br + 16 + half * 8);
      v16bf bfr;
#pragma unroll
      for (int t = 0; t < 8; ++t) { bfr[t] = blo[t]; bfr[8 + t] = bhi[t]; }
#pragma unroll
      for (int s = 0; s < 2; ++s)
        acc[s][c] = __builtin_amdgcn_wmma_f32_16x16x32_bf16(
            false, afr[s], false, bfr, (short)0, acc[s][c], false, false);
    }
    __syncthreads();               // all reads of `cur` done before reuse
  }

#pragma unroll
  for (int s = 0; s < 2; ++s)
#pragma unroll
    for (int c = 0; c < 4; ++c) {
      const int col = n0 + nw * 64 + c * 16 + ln;  // C layout: lane = column
      float bv = (EPI >= 1) ? bias[col] : 0.0f;
#pragma unroll
      for (int r = 0; r < 8; ++r) {                // VGPR r = row half*8+r
        int row = m0 + mw * 32 + s * 16 + half * 8 + r;
        float v = acc[s][c][r];
        if (EPI >= 1) v += bv;
        if (EPI == 2) v = 0.5f * v * (1.0f + erff(v * 0.70710678118654752f));
        size_t idx = (size_t)row * N + col;
        if (OUTBF) Cb[idx] = f2bf(v);
        else       Cf[idx] = v;
      }
    }
}

// ---------------------------------------------------------------------------
// Causal flash attention: one wave per (batch, head, 16-row q tile).
// Streams 32 keys/iter: 4 WMMAs QK^T, online softmax, 4 WMMAs P@V.
// Q/K/V/ctx are [NTOK, D_MODEL] bf16 with head h at columns [h*64, h*64+64).
// ---------------------------------------------------------------------------
__global__ void __launch_bounds__(32)
flash_attn_wmma(const __bf16* __restrict__ Q, const __bf16* __restrict__ K,
                const __bf16* __restrict__ V, __bf16* __restrict__ ctx) {
  const int lane = threadIdx.x & 31;
  const int half = lane >> 4;
  const int ln   = lane & 15;
  const int qt = blockIdx.x, h = blockIdx.y, b = blockIdx.z;
  const int q0 = qt * 16;
  const size_t rs = D_MODEL;

  const __bf16* Qb = Q + (size_t)(b * SEQ + q0) * rs + h * DK;
  const __bf16* Kb = K + (size_t)(b * SEQ) * rs + h * DK;
  const __bf16* Vb = V + (size_t)(b * SEQ) * rs + h * DK;

  const __bf16* qrow = Qb + (size_t)ln * rs;
  v16bf aq0, aq1;
  {
    v8bf l0 = *(const v8bf*)(qrow + half * 16);
    v8bf h0 = *(const v8bf*)(qrow + half * 16 + 8);
    v8bf l1 = *(const v8bf*)(qrow + 32 + half * 16);
    v8bf h1 = *(const v8bf*)(qrow + 40 + half * 16);
#pragma unroll
    for (int t = 0; t < 8; ++t) {
      aq0[t] = l0[t]; aq0[8 + t] = h0[t];
      aq1[t] = l1[t]; aq1[8 + t] = h1[t];
    }
  }

  v8f zero = {};
  v8f o[4];
#pragma unroll
  for (int g = 0; g < 4; ++g) o[g] = zero;
  float mrow[8], lrow[8];
#pragma unroll
  for (int r = 0; r < 8; ++r) { mrow[r] = -3.0e38f; lrow[r] = 0.0f; }

  __shared__ __align__(64) __bf16 Plds[16][32];

  const int kend = q0 + 16;                     // causal: keys 0..q0+15
  for (int k0 = 0; k0 < kend; k0 += 32) {
    v8f sc[2];
#pragma unroll
    for (int j = 0; j < 2; ++j) {
      const __bf16* krow = Kb + (size_t)(k0 + j * 16 + ln) * rs;
      v16bf bk0 = *(const v16bf*)(krow + half * 16);
      v16bf bk1 = *(const v16bf*)(krow + 32 + half * 16);
      v8f c = zero;
      c = __builtin_amdgcn_wmma_f32_16x16x32_bf16(false, aq0, false, bk0,
                                                  (short)0, c, false, false);
      c = __builtin_amdgcn_wmma_f32_16x16x32_bf16(false, aq1, false, bk1,
                                                  (short)0, c, false, false);
      sc[j] = c;
    }
#pragma unroll
    for (int j = 0; j < 2; ++j) {
      int colg = k0 + j * 16 + ln;
#pragma unroll
      for (int r = 0; r < 8; ++r) {
        int rowg = q0 + half * 8 + r;
        float v = sc[j][r] * 0.125f;            // 1/sqrt(64)
        sc[j][r] = (colg <= rowg) ? v : -3.0e38f;
      }
    }
#pragma unroll
    for (int r = 0; r < 8; ++r) {
      float rm = fmaxf(sc[0][r], sc[1][r]);
#pragma unroll
      for (int off = 1; off < 16; off <<= 1)
        rm = fmaxf(rm, __shfl_xor(rm, off, 32));
      float mnew  = fmaxf(mrow[r], rm);
      float alpha = __expf(mrow[r] - mnew);
      mrow[r] = mnew;
      float p0 = __expf(sc[0][r] - mnew);
      float p1 = __expf(sc[1][r] - mnew);
      Plds[half * 8 + r][ln]      = f2bf(p0);
      Plds[half * 8 + r][16 + ln] = f2bf(p1);
      float rsum = p0 + p1;
#pragma unroll
      for (int off = 1; off < 16; off <<= 1)
        rsum += __shfl_xor(rsum, off, 32);
      lrow[r] = lrow[r] * alpha + rsum;
#pragma unroll
      for (int g = 0; g < 4; ++g) o[g][r] *= alpha;
    }
    __syncthreads();
    v16bf ap;
    {
      v8bf plo = *(const v8bf*)&Plds[ln][half * 8];
      v8bf phi = *(const v8bf*)&Plds[ln][16 + half * 8];
#pragma unroll
      for (int t = 0; t < 8; ++t) { ap[t] = plo[t]; ap[8 + t] = phi[t]; }
    }
#pragma unroll
    for (int g = 0; g < 4; ++g) {
      v16bf bv;
#pragma unroll
      for (int t = 0; t < 16; ++t)
        bv[t] = Vb[(size_t)(k0 + half * 8 + t + (t & 8)) * rs + g * 16 + ln];
      o[g] = __builtin_amdgcn_wmma_f32_16x16x32_bf16(false, ap, false, bv,
                                                     (short)0, o[g], false, false);
    }
    __syncthreads();
  }
#pragma unroll
  for (int r = 0; r < 8; ++r) {
    float inv = 1.0f / lrow[r];
    int rowg = b * SEQ + q0 + half * 8 + r;
#pragma unroll
    for (int g = 0; g < 4; ++g)
      ctx[(size_t)rowg * rs + h * DK + g * 16 + ln] = f2bf(o[g][r] * inv);
  }
}

// ---------------------------------------------------------------------------
// out = LayerNorm(X + Y) * gamma + beta ; optional bf16 copy for next GEMM.
// ---------------------------------------------------------------------------
__global__ void __launch_bounds__(256)
add_layernorm(const float* __restrict__ X, const float* __restrict__ Y,
              const float* __restrict__ gamma, const float* __restrict__ beta,
              float* __restrict__ outF, __bf16* __restrict__ outB) {
  const int row = blockIdx.x;
  const int tid = threadIdx.x;
  const size_t base = (size_t)row * D_MODEL;
  const int c0 = tid * 4;

  float4 xa = *(const float4*)(X + base + c0);
  float4 ya = *(const float4*)(Y + base + c0);
  float v[4] = {xa.x + ya.x, xa.y + ya.y, xa.z + ya.z, xa.w + ya.w};
  float s = 0.f, s2 = 0.f;
#pragma unroll
  for (int i = 0; i < 4; ++i) { s += v[i]; s2 += v[i] * v[i]; }

  __shared__ float redS[8], redS2[8];
  const int lane = tid & 31, wave = tid >> 5;
#pragma unroll
  for (int off = 1; off < 32; off <<= 1) {
    s  += __shfl_xor(s, off, 32);
    s2 += __shfl_xor(s2, off, 32);
  }
  if (lane == 0) { redS[wave] = s; redS2[wave] = s2; }
  __syncthreads();
  if (tid == 0) {
    float ts = 0.f, ts2 = 0.f;
#pragma unroll
    for (int w = 0; w < 8; ++w) { ts += redS[w]; ts2 += redS2[w]; }
    redS[0] = ts; redS2[0] = ts2;
  }
  __syncthreads();
  const float mean = redS[0] * (1.0f / D_MODEL);
  const float var  = redS2[0] * (1.0f / D_MODEL) - mean * mean;
  const float rstd = rsqrtf(var + 1e-5f);
#pragma unroll
  for (int i = 0; i < 4; ++i) {
    int c = c0 + i;
    float o = (v[i] - mean) * rstd * gamma[c] + beta[c];
    outF[base + c] = o;
    if (outB) outB[base + c] = f2bf(o);
  }
}

// ---------------------------------------------------------------------------
extern "C" void kernel_launch(void* const* d_in, const int* in_sizes, int n_in,
                              void* d_out, int out_size, void* d_ws,
                              size_t ws_size, hipStream_t stream) {
  (void)in_sizes; (void)n_in; (void)out_size; (void)ws_size;
  const float* x   = (const float*)d_in[0];
  const float* Wq  = (const float*)d_in[1];
  const float* Wk  = (const float*)d_in[2];
  const float* Wv  = (const float*)d_in[3];
  const float* Wo  = (const float*)d_in[4];
  const float* g1  = (const float*)d_in[5];
  const float* b1  = (const float*)d_in[6];
  const float* g2  = (const float*)d_in[7];
  const float* b2  = (const float*)d_in[8];
  const float* W1  = (const float*)d_in[9];
  const float* bf1 = (const float*)d_in[10];
  const float* W2  = (const float*)d_in[11];
  const float* bf2 = (const float*)d_in[12];
  float* out = (float*)d_out;

  char* ws = (char*)d_ws;
  size_t off = 0;
  auto alloc = [&](size_t bytes) -> void* {
    void* p = ws + off;
    off = (off + bytes + 255) & ~(size_t)255;
    return p;
  };
  __bf16* xb   = (__bf16*)alloc((size_t)NTOK * D_MODEL * 2);
  __bf16* Wqb  = (__bf16*)alloc((size_t)D_MODEL * D_MODEL * 2);
  __bf16* Wkb  = (__bf16*)alloc((size_t)D_MODEL * D_MODEL * 2);
  __bf16* Wvb  = (__bf16*)alloc((size_t)D_MODEL * D_MODEL * 2);
  __bf16* Wob  = (__bf16*)alloc((size_t)D_MODEL * D_MODEL * 2);
  __bf16* W1b  = (__bf16*)alloc((size_t)D_FF * D_MODEL * 2);
  __bf16* W2b  = (__bf16*)alloc((size_t)D_MODEL * D_FF * 2);
  __bf16* Qb   = (__bf16*)alloc((size_t)NTOK * D_MODEL * 2);
  __bf16* Kb   = (__bf16*)alloc((size_t)NTOK * D_MODEL * 2);
  __bf16* Vb   = (__bf16*)alloc((size_t)NTOK * D_MODEL * 2);
  __bf16* ctxb = (__bf16*)alloc((size_t)NTOK * D_MODEL * 2);
  float*  attn = (float*)alloc((size_t)NTOK * D_MODEL * 4);
  float*  hres = (float*)alloc((size_t)NTOK * D_MODEL * 4);
  __bf16* hb   = (__bf16*)alloc((size_t)NTOK * D_MODEL * 2);
  __bf16* f1b  = (__bf16*)alloc((size_t)NTOK * D_FF * 2);
  float*  f2o  = (float*)alloc((size_t)NTOK * D_MODEL * 4);

  auto cvt = [&](const float* src, __bf16* dst, int n) {
    cvt_f32_to_bf16<<<(n + 255) / 256, 256, 0, stream>>>(src, dst, n);
  };
  cvt(x,  xb,  NTOK * D_MODEL);
  cvt(Wq, Wqb, D_MODEL * D_MODEL);
  cvt(Wk, Wkb, D_MODEL * D_MODEL);
  cvt(Wv, Wvb, D_MODEL * D_MODEL);
  cvt(Wo, Wob, D_MODEL * D_MODEL);
  cvt(W1, W1b, D_FF * D_MODEL);
  cvt(W2, W2b, D_MODEL * D_FF);

  dim3 blk(256);
  dim3 gProj(D_MODEL / BN, NTOK / BM);     // 128x128 block tiles
  gemm_wmma_bf16<0, true><<<gProj, blk, 0, stream>>>(
      xb, Wqb, nullptr, nullptr, Qb, NTOK, D_MODEL, D_MODEL);
  gemm_wmma_bf16<0, true><<<gProj, blk, 0, stream>>>(
      xb, Wkb, nullptr, nullptr, Kb, NTOK, D_MODEL, D_MODEL);
  gemm_wmma_bf16<0, true><<<gProj, blk, 0, stream>>>(
      xb, Wvb, nullptr, nullptr, Vb, NTOK, D_MODEL, D_MODEL);

  flash_attn_wmma<<<dim3(SEQ / 16, NUM_HEADS, BATCH), dim3(32), 0, stream>>>(
      Qb, Kb, Vb, ctxb);

  gemm_wmma_bf16<0, false><<<gProj, blk, 0, stream>>>(
      ctxb, Wob, nullptr, attn, nullptr, NTOK, D_MODEL, D_MODEL);

  add_layernorm<<<NTOK, 256, 0, stream>>>(x, attn, g1, b1, hres, hb);

  gemm_wmma_bf16<2, true><<<dim3(D_FF / BN, NTOK / BM), blk, 0, stream>>>(
      hb, W1b, bf1, nullptr, f1b, NTOK, D_FF, D_MODEL);
  gemm_wmma_bf16<1, false><<<gProj, blk, 0, stream>>>(
      f1b, W2b, bf2, f2o, nullptr, NTOK, D_MODEL, D_FF);

  add_layernorm<<<NTOK, 256, 0, stream>>>(hres, f2o, g2, b2, out, nullptr);
}
